// BailingMoeV2_5LinearAttention_2362232012862
// MI455X (gfx1250) — matrix operand order repaired and checked by reading, not compiled, not run on hardware
//
#include <hip/hip_runtime.h>
#include <hip/hip_bf16.h>

// ---------------------------------------------------------------------------
// Types for CDNA5 WMMA
// ---------------------------------------------------------------------------
typedef __attribute__((ext_vector_type(16))) __bf16        v16bf;
typedef __attribute__((ext_vector_type(2)))  __bf16        v2bf;
typedef __attribute__((ext_vector_type(8)))  float         v8f;
typedef __attribute__((ext_vector_type(8)))  unsigned int  v8u;

__device__ __forceinline__ v8f zero8() {
  v8f z;
#pragma unroll
  for (int j = 0; j < 8; ++j) z[j] = 0.0f;
  return z;
}

// --- native bf16 conversion (hardware v_cvt on gfx1250; RTE either way) ----
__device__ __forceinline__ unsigned short f2bfu(float f) {
  return __builtin_bit_cast(unsigned short, (__bf16)f);
}
__device__ __forceinline__ float bfu2f(unsigned short h) {
  unsigned u = ((unsigned)h) << 16;
  return __builtin_bit_cast(float, u);
}
__device__ __forceinline__ unsigned packbf(float a, float b) {
#if __has_builtin(__builtin_amdgcn_cvt_pk_bf16_f32)
  return __builtin_bit_cast(unsigned, __builtin_amdgcn_cvt_pk_bf16_f32(a, b));
#else
  v2bf t;
  t[0] = (__bf16)a;
  t[1] = (__bf16)b;
  return __builtin_bit_cast(unsigned, t);
#endif
}

__device__ __forceinline__ v8f wmma_bf16(v8u a, v8u b, v8f c) {
  return __builtin_amdgcn_wmma_f32_16x16x32_bf16(
      false, __builtin_bit_cast(v16bf, a),
      false, __builtin_bit_cast(v16bf, b),
      (short)0, c, false, false);
}

// ---------------------------------------------------------------------------
// Async global -> LDS (CDNA5, ASYNCcnt-tracked).  VDST = LDS byte offset,
// VADDR = 64-bit global address, no SADDR.  LDS offset = low 32 bits of the
// generic pointer (flat->LDS mapping truncates, ISA 10.2).
// ---------------------------------------------------------------------------
__device__ __forceinline__ void async_copy16(void* ldsdst, const void* gsrc) {
  unsigned lds_off = (unsigned)(unsigned long long)(uintptr_t)ldsdst;
  unsigned long long gaddr = (unsigned long long)(uintptr_t)gsrc;
  asm volatile("global_load_async_to_lds_b128 %0, %1, off"
               :: "v"(lds_off), "v"(gaddr)
               : "memory");
}
__device__ __forceinline__ void async_wait0() {
  asm volatile("s_wait_asynccnt 0x0" ::: "memory");
}

// ---------------------------------------------------------------------------
// Fragment loaders (A/B: contiguous k order used by this code; C/D layout per CDNA5 ISA 7.12.2)
// A 16x32 bf16: lane = m + 16*(k>>4);     vgpr = (k&15)>>1;             half = k&1
// B 32x16 bf16: lane = n + 16*(k>>4);     vgpr = (k&15)>>1;             half = k&1
// C/D 16x16 f32: lane = n + 16*(m>>3);    vgpr = m&7
// ---------------------------------------------------------------------------
__device__ __forceinline__ v8u frag_a(const unsigned short* p, int lda, int row0, int k0) {
  int lane = threadIdx.x & 31;
  int m = lane & 15, hi = lane >> 4;
  const unsigned short* rowp = p + (row0 + m) * lda + k0;
  v8u r;
#pragma unroll
  for (int j = 0; j < 8; ++j) {
    int kb = 2 * j + 16 * hi;
    r[j] = *(const unsigned int*)(rowp + kb);
  }
  return r;
}

// B operand from LDS stored "n-major": lds[n*ldb + k]
__device__ __forceinline__ v8u frag_b(const unsigned short* p, int ldb, int n0, int k0) {
  int lane = threadIdx.x & 31;
  int n = lane & 15, hi = lane >> 4;
  const unsigned short* colp = p + (n0 + n) * ldb + k0 + 16 * hi;
  v8u r;
#pragma unroll
  for (int j = 0; j < 8; ++j) r[j] = *(const unsigned int*)(colp + 2 * j);
  return r;
}

// A operand with a per-row (time) scale applied (row is uniform per lane)
__device__ __forceinline__ v8u frag_a_rowscale(const unsigned short* p, int lda, int row0,
                                               int k0, float w) {
  v8u r = frag_a(p, lda, row0, k0);
#pragma unroll
  for (int j = 0; j < 8; ++j) {
    unsigned u = r[j];
    r[j] = packbf(bfu2f((unsigned short)(u & 0xffff)) * w,
                  bfu2f((unsigned short)(u >> 16)) * w);
  }
  return r;
}

// B operand from fp32 LDS (S state): lds[k*ld + n], converted to bf16 on the fly
__device__ __forceinline__ v8u frag_b_f32(const float* p, int ld, int n0, int k0) {
  int lane = threadIdx.x & 31;
  int n = lane & 15, hi = lane >> 4;
  v8u r;
#pragma unroll
  for (int j = 0; j < 8; ++j) {
    int k = k0 + 2 * j + 16 * hi;
    r[j] = packbf(p[k * ld + n0 + n], p[(k + 1) * ld + n0 + n]);
  }
  return r;
}

// A operand = Ktilde^T: element (m=dk, k=time j) = K[j][dk] * dpow[63-j]
__device__ __forceinline__ v8u frag_a_ktw(const unsigned short* Ks, int dk0, int k0,
                                          const float* dpow) {
  int lane = threadIdx.x & 31;
  int m = dk0 + (lane & 15), hi = lane >> 4;
  v8u r;
#pragma unroll
  for (int j = 0; j < 8; ++j) {
    int kb = k0 + 2 * j + 16 * hi;
    float f0 = bfu2f(Ks[kb * 128 + m]) * dpow[63 - kb];
    float f1 = bfu2f(Ks[(kb + 1) * 128 + m]) * dpow[62 - kb];
    r[j] = packbf(f0, f1);
  }
  return r;
}

// ---------------------------------------------------------------------------
// fp32 -> bf16 conversion
// ---------------------------------------------------------------------------
__global__ void to_bf16_kernel(const float* __restrict__ in, unsigned short* __restrict__ out,
                               int n) {
  int i = blockIdx.x * blockDim.x + threadIdx.x;
  int stride = gridDim.x * blockDim.x;
  for (; i < n; i += stride) out[i] = f2bfu(in[i]);
}

// ---------------------------------------------------------------------------
// Tiled bf16 WMMA GEMM: C[M,N] = A[M,K] * B[K,N] (both row-major bf16)
// 256 threads, 128x128 block tile, K-step 32. 8 waves: 2x4 tiles of 16x16 each.
// EPI 0: store f32.  EPI 1: sigmoid -> f32.  EPI 2: qkv split (q f32, k f32, v bf16)
// ---------------------------------------------------------------------------
template <int EPI>
__global__ __launch_bounds__(256) void gemm_bf16_kernel(
    const unsigned short* __restrict__ A, const unsigned short* __restrict__ B,
    int M, int N, int K, float* __restrict__ out0, float* __restrict__ out1,
    unsigned short* __restrict__ out2) {
  __shared__ __align__(16) unsigned short As[128 * 32];
  __shared__ __align__(16) unsigned short Bt[128 * 32];  // transposed: Bt[n][k]

  int m0 = blockIdx.y * 128, n0 = blockIdx.x * 128;
  int tid = threadIdx.x;
  int w = tid >> 5;
  int wr = w >> 1, wc = w & 1;  // wave tile: rows [wr*32, +32), cols [wc*64, +64)

  v8f acc[2][4];
#pragma unroll
  for (int rt = 0; rt < 2; ++rt)
#pragma unroll
    for (int ct = 0; ct < 4; ++ct) acc[rt][ct] = zero8();

  for (int kk = 0; kk < K; kk += 32) {
    // ---- stage A tile (128x32) via async global->LDS DMA (ASYNCcnt)
#pragma unroll
    for (int i = 0; i < 2; ++i) {
      int idx = tid * 2 + i;          // 0..511
      int row = idx >> 2;             // 4 x 16B per row
      int c8 = (idx & 3) * 8;         // halfword offset
      async_copy16(As + row * 32 + c8, A + (size_t)(m0 + row) * K + kk + c8);
    }
    // ---- stage B tile (32x128) transposed into Bt[n][k]
#pragma unroll
    for (int i = 0; i < 8; ++i) {
      int p = tid + i * 256;          // 0..2047 (pairs over k)
      int n = p & 127;
      int k = (p >> 7) * 2;
      unsigned b0 = B[(size_t)(kk + k) * N + n0 + n];
      unsigned b1 = B[(size_t)(kk + k + 1) * N + n0 + n];
      *(unsigned*)(Bt + n * 32 + k) = b0 | (b1 << 16);
    }
    async_wait0();
    __syncthreads();

    if (kk + 32 < K) {  // gfx1250 global_prefetch_b8 for the next K tile
      __builtin_prefetch(A + (size_t)(m0 + (tid >> 1)) * K + kk + 32, 0, 1);
      __builtin_prefetch(B + (size_t)(kk + 32 + (tid & 31)) * N + n0 + ((tid >> 5) << 4), 0, 1);
    }

    v8u af[2];
    af[0] = frag_a(As, 32, wr * 32, 0);
    af[1] = frag_a(As, 32, wr * 32 + 16, 0);
    v8u bfr[4];
#pragma unroll
    for (int ct = 0; ct < 4; ++ct) bfr[ct] = frag_b(Bt, 32, wc * 64 + ct * 16, 0);
#pragma unroll
    for (int rt = 0; rt < 2; ++rt)
#pragma unroll
      for (int ct = 0; ct < 4; ++ct) acc[rt][ct] = wmma_bf16(af[rt], bfr[ct], acc[rt][ct]);
    __syncthreads();
  }

  // ---- epilogue (C layout: lane holds m = v + 8*(lane>>4), n = lane&15)
  int lane = tid & 31, nn = lane & 15, hi = lane >> 4;
#pragma unroll
  for (int rt = 0; rt < 2; ++rt)
#pragma unroll
    for (int ct = 0; ct < 4; ++ct)
#pragma unroll
      for (int v = 0; v < 8; ++v) {
        int r = m0 + wr * 32 + rt * 16 + v + 8 * hi;
        int c = n0 + wc * 64 + ct * 16 + nn;
        float val = acc[rt][ct][v];
        if (EPI == 0) {
          out0[(size_t)r * N + c] = val;
        } else if (EPI == 1) {
          out0[(size_t)r * N + c] = 1.0f / (1.0f + __expf(-val));
        } else {
          int which = c >> 11;      // col / 2048
          int rem = c & 2047;
          if (which == 0)      out0[(size_t)r * 2048 + rem] = val;
          else if (which == 1) out1[(size_t)r * 2048 + rem] = val;
          else                 out2[(size_t)r * 2048 + rem] = f2bfu(val);
        }
      }
}

// ---------------------------------------------------------------------------
// RMS-norm + RoPE + scale for q and k. One block per (t,h), 128 threads.
// ---------------------------------------------------------------------------
__global__ __launch_bounds__(128) void qk_post_kernel(
    const float* __restrict__ qf, const float* __restrict__ kf,
    const int* __restrict__ positions, const float* __restrict__ qnw,
    const float* __restrict__ knw, unsigned short* __restrict__ qb,
    unsigned short* __restrict__ kb) {
  __shared__ float red[128];
  __shared__ float xn[128];
  int t = blockIdx.x, h = blockIdx.y, d = threadIdx.x;
  float pos = (float)positions[t];
  size_t base = (size_t)t * 2048 + h * 128 + d;

  for (int which = 0; which < 2; ++which) {
    const float* src = which ? kf : qf;
    const float* wv = which ? knw : qnw;
    unsigned short* dst = which ? kb : qb;

    float v = src[base];
    red[d] = v * v;
    __syncthreads();
    for (int s2 = 64; s2 > 0; s2 >>= 1) {
      if (d < s2) red[d] += red[d + s2];
      __syncthreads();
    }
    float inv = rsqrtf(red[0] / 128.0f + 1e-6f);
    float nv = v * inv * wv[d];
    xn[d] = nv;
    __syncthreads();

    float out;
    if (d < 64) {
      int i = d & 31;
      // inv_freq = 10000^(-i/32) = exp(-(ln 1e4)/32 * i)
      float fr = pos * __expf(-0.28782313662425572f * (float)i);
      float sn, cs;
      __sincosf(fr, &sn, &cs);
      float x1 = xn[d & 31];
      float x2 = xn[(d & 31) + 32];
      out = (d < 32) ? (x1 * cs - x2 * sn) : (x2 * cs + x1 * sn);
    } else {
      out = nv;
    }
    if (which == 0) out *= 0.08838834764831845f;  // D^-0.5
    dst[base] = f2bfu(out);
    __syncthreads();
  }
}

// ---------------------------------------------------------------------------
// Chunked linear-attention scan.  One block per (head, dv-slice of 32).
// S (128 x 32 fp32) lives in LDS across the 64 sequential chunks of C=64.
// ---------------------------------------------------------------------------
__global__ __launch_bounds__(256) void scan_kernel(
    const unsigned short* __restrict__ qb, const unsigned short* __restrict__ kb,
    const unsigned short* __restrict__ vb, const float* __restrict__ rstate,
    float* __restrict__ attn, float* __restrict__ sfinal) {
  __shared__ __align__(16) float S[128 * 32];           // [dk][dv_local]
  __shared__ __align__(16) unsigned short Qs[64 * 128]; // [time][dk]
  __shared__ __align__(16) unsigned short Ks[64 * 128]; // [time][dk]
  __shared__ __align__(16) unsigned short Vt[32 * 64];  // [dv_local][time]
  __shared__ __align__(16) unsigned short Am[64 * 64];  // masked decayed QK^T
  __shared__ float dpow[72];

  int h = blockIdx.x >> 2;
  int dv0 = (blockIdx.x & 3) * 32;
  int tid = threadIdx.x;

  // per-head decay: slope = -2^(-0.5*(h+1)) * (1 - 1/31 + 1e-5), lambda = exp(slope)
  float slope = -exp2f(-0.5f * (float)(h + 1)) * (1.0f - (1.0f / 31.0f) + 1e-5f);
  if (tid < 65) dpow[tid] = __expf(slope * (float)tid);

  for (int i = tid; i < 128 * 32; i += 256) {
    int dk = i >> 5, dvl = i & 31;
    S[dk * 32 + dvl] = rstate[(size_t)h * 16384 + dk * 128 + dv0 + dvl];
  }
  __syncthreads();

  int w = tid >> 5, lane = tid & 31;
  int nn = lane & 15, hi = lane >> 4;

  for (int cc = 0; cc < 64; ++cc) {
    int t0 = cc * 64;
    // ---- stage Q, K (64x128 bf16 each) via async global->LDS DMA
#pragma unroll
    for (int i = 0; i < 4; ++i) {
      int idx = tid + i * 256;     // 0..1023
      int row = idx >> 4;          // 16 x 16B per row
      int c8 = (idx & 15) * 8;
      async_copy16(Qs + row * 128 + c8, qb + (size_t)(t0 + row) * 2048 + h * 128 + c8);
      async_copy16(Ks + row * 128 + c8, kb + (size_t)(t0 + row) * 2048 + h * 128 + c8);
    }
    // ---- stage V transposed: Vt[dv][time]
#pragma unroll
    for (int i = 0; i < 4; ++i) {
      int p = tid + i * 256;       // 0..1023 (pairs over time)
      int dv = p & 31;
      int j = (p >> 5) * 2;
      unsigned v0 = vb[(size_t)(t0 + j) * 2048 + h * 128 + dv0 + dv];
      unsigned v1 = vb[(size_t)(t0 + j + 1) * 2048 + h * 128 + dv0 + dv];
      *(unsigned*)(Vt + dv * 64 + j) = v0 | (v1 << 16);
    }
    async_wait0();
    __syncthreads();

    // ---- phase 1: Am = mask(Q K^T) * lambda^(t-s)
    {
      int rt = w >> 1;
      int ctb = (w & 1) * 2;
#pragma unroll
      for (int cti = 0; cti < 2; ++cti) {
        int ct = ctb + cti;
        v8f acc = zero8();
#pragma unroll
        for (int ks = 0; ks < 128; ks += 32)
          acc = wmma_bf16(frag_a(Qs, 128, rt * 16, ks), frag_b(Ks, 128, ct * 16, ks), acc);
#pragma unroll
        for (int v = 0; v < 8; ++v) {
          int tl = rt * 16 + v + 8 * hi;
          int sl = ct * 16 + nn;
          float val = (sl <= tl) ? acc[v] * dpow[tl - sl] : 0.0f;
          Am[tl * 64 + sl] = f2bfu(val);
        }
      }
    }
    __syncthreads();

    // ---- phase 2: o = Am*V + (lambda^(t+1) Q) * S   -> attn
    {
      int rt = w >> 1;
      int ct = w & 1;
      v8f acc = zero8();
#pragma unroll
      for (int ks = 0; ks < 64; ks += 32)
        acc = wmma_bf16(frag_a(Am, 64, rt * 16, ks), frag_b(Vt, 64, ct * 16, ks), acc);
      float wq = dpow[rt * 16 + nn + 1];  // lane row = rt*16 + (lane&15)
#pragma unroll
      for (int ks = 0; ks < 128; ks += 32)
        acc = wmma_bf16(frag_a_rowscale(Qs, 128, rt * 16, ks, wq),
                        frag_b_f32(S, 32, ct * 16, ks), acc);
#pragma unroll
      for (int v = 0; v < 8; ++v) {
        int tg = t0 + rt * 16 + v + 8 * hi;
        attn[(size_t)tg * 2048 + h * 128 + dv0 + ct * 16 + nn] = acc[v];
      }
    }
    __syncthreads();

    // ---- phase 3: S = lambda^64 * S + Ktilde^T V   (each wave owns 16 dk rows)
    {
      int dk0 = w * 16;
      float dC = dpow[64];
#pragma unroll
      for (int ct = 0; ct < 2; ++ct) {
        v8f c = zero8();
#pragma unroll
        for (int v = 0; v < 8; ++v)
          c[v] = S[(dk0 + v + 8 * hi) * 32 + ct * 16 + nn] * dC;
#pragma unroll
        for (int ks = 0; ks < 64; ks += 32)
          c = wmma_bf16(frag_a_ktw(Ks, dk0, ks, dpow), frag_b(Vt, 64, ct * 16, ks), c);
#pragma unroll
        for (int v = 0; v < 8; ++v)
          S[(dk0 + v + 8 * hi) * 32 + ct * 16 + nn] = c[v];
      }
    }
    __syncthreads();
  }

  // ---- final state
  for (int i = tid; i < 128 * 32; i += 256) {
    int dk = i >> 5, dvl = i & 31;
    sfinal[(size_t)h * 16384 + dk * 128 + dv0 + dvl] = S[dk * 32 + dvl];
  }
}

// ---------------------------------------------------------------------------
// Group RMS-norm (8 groups of 256 == one wave each) * g_norm_w * gate -> bf16
// ---------------------------------------------------------------------------
__global__ __launch_bounds__(256) void gnorm_gate_kernel(
    const float* __restrict__ attn, const float* __restrict__ gate,
    const float* __restrict__ gnw, unsigned short* __restrict__ xb) {
  int t = blockIdx.x, tid = threadIdx.x;
  size_t base = (size_t)t * 2048 + tid * 8;
  float x[8];
  float ss = 0.0f;
#pragma unroll
  for (int j = 0; j < 8; ++j) {
    x[j] = attn[base + j];
    ss += x[j] * x[j];
  }
#pragma unroll
  for (int off = 16; off > 0; off >>= 1) ss += __shfl_xor(ss, off);
  float r = rsqrtf(ss / 256.0f + 1e-6f);
#pragma unroll
  for (int j = 0; j < 8; ++j)
    xb[base + j] = f2bfu(x[j] * r * gnw[tid * 8 + j] * gate[base + j]);
}

// ---------------------------------------------------------------------------
// Host-side orchestration
// ---------------------------------------------------------------------------
extern "C" void kernel_launch(void* const* d_in, const int* in_sizes, int n_in,
                              void* d_out, int out_size, void* d_ws, size_t ws_size,
                              hipStream_t stream) {
  (void)in_sizes; (void)n_in; (void)out_size; (void)ws_size;
  const int*   positions = (const int*)d_in[0];
  const float* hidden    = (const float*)d_in[1];
  const float* rstate    = (const float*)d_in[2];
  const float* w_qkv     = (const float*)d_in[3];
  const float* w_g       = (const float*)d_in[4];
  const float* w_dense   = (const float*)d_in[5];
  const float* qnw       = (const float*)d_in[6];
  const float* knw       = (const float*)d_in[7];
  const float* gnw       = (const float*)d_in[8];
  float* out = (float*)d_out;

  char* ws = (char*)d_ws;
  size_t off = 0;
  auto carve = [&](size_t bytes) -> void* {
    void* p = ws + off;
    off += (bytes + 255) & ~(size_t)255;
    return p;
  };
  unsigned short* hb    = (unsigned short*)carve(4096ull * 2048 * 2);
  unsigned short* wqkvb = (unsigned short*)carve(2048ull * 6144 * 2);
  unsigned short* wgb   = (unsigned short*)carve(2048ull * 2048 * 2);
  unsigned short* wdb   = (unsigned short*)carve(2048ull * 2048 * 2);
  float* qf = (float*)carve(4096ull * 2048 * 4);
  float* kf = (float*)carve(4096ull * 2048 * 4);
  unsigned short* vb  = (unsigned short*)carve(4096ull * 2048 * 2);
  unsigned short* qb2 = (unsigned short*)carve(4096ull * 2048 * 2);
  unsigned short* kb2 = (unsigned short*)carve(4096ull * 2048 * 2);
  // aliases (lifetimes do not overlap in stream order)
  float* gate = qf;           // written after qk_post consumed qf
  float* attn = kf;           // written after qk_post consumed kf
  unsigned short* xb = vb;    // written after scan consumed vb

  to_bf16_kernel<<<1024, 256, 0, stream>>>(hidden, hb, 4096 * 2048);
  to_bf16_kernel<<<1024, 256, 0, stream>>>(w_qkv, wqkvb, 2048 * 6144);
  to_bf16_kernel<<<1024, 256, 0, stream>>>(w_g, wgb, 2048 * 2048);
  to_bf16_kernel<<<1024, 256, 0, stream>>>(w_dense, wdb, 2048 * 2048);

  // qkv = hidden @ w_qkv  (split epilogue: q f32, k f32, v bf16)
  gemm_bf16_kernel<2><<<dim3(48, 32), 256, 0, stream>>>(hb, wqkvb, 4096, 6144, 2048,
                                                        qf, kf, vb);
  // RMS + RoPE + scale -> bf16 q/k
  qk_post_kernel<<<dim3(4096, 16), 128, 0, stream>>>(qf, kf, positions, qnw, knw, qb2, kb2);
  // gate = sigmoid(hidden @ w_g)  (reuses qf storage)
  gemm_bf16_kernel<1><<<dim3(16, 32), 256, 0, stream>>>(hb, wgb, 4096, 2048, 2048,
                                                        gate, nullptr, nullptr);
  // chunked linear-attention scan (16 heads x 4 dv-slices)
  scan_kernel<<<64, 256, 0, stream>>>(qb2, kb2, vb, rstate, attn, out + 4096ull * 2048);
  // group norm * gate -> bf16
  gnorm_gate_kernel<<<4096, 256, 0, stream>>>(attn, gate, gnw, xb);
  // out = x @ w_dense
  gemm_bf16_kernel<0><<<dim3(16, 32), 256, 0, stream>>>(xb, wdb, 4096, 2048, 2048,
                                                        out, nullptr, nullptr);
}
